// KANLinear_73186242723913
// MI455X (gfx1250) — matrix operand order repaired and checked by reading, not compiled, not run on hardware
//
#include <hip/hip_runtime.h>
#include <hip/hip_bf16.h>

typedef _Float16 half_t;
typedef __attribute__((ext_vector_type(16))) _Float16 v16h;
typedef __attribute__((ext_vector_type(8)))  _Float16 v8h;
typedef __attribute__((ext_vector_type(4)))  _Float16 v4h;
typedef __attribute__((ext_vector_type(8)))  float    v8f;
typedef __attribute__((ext_vector_type(4)))  float    v4f;

#define B_DIM   8192
#define IN_DIM  1024
#define OUT_DIM 1024
#define COEFF   8

#define BM 128
#define BN 128
#define KP 64              // halfs of packed K per panel
#define KPAD (KP + 8)      // padded LDS row stride (144 B -> conflict-free frag reads)
#define NTHREADS 256
#define NPA (IN_DIM / KP)           // 16 phase-A panels
#define NPB (IN_DIM / (KP / COEFF)) // 128 phase-B panels (8 inputs each)

__device__ __forceinline__ float silu_f(float v) {
    return v * (1.0f / (1.0f + __expf(-v)));
}

// Cubic B-spline bases, knots t[j] = (j-3)*0.2, j=0..11 (uniform ->
// constant denominators). Mirrors the reference Cox-de-Boor recursion.
__device__ __forceinline__ void eval_bases(float xv, float bb[COEFF]) {
    const float h = 0.2f;
    float b[11];
#pragma unroll
    for (int j = 0; j < 11; ++j) {
        float t0 = (float)(j - 3) * h;
        float t1 = (float)(j - 2) * h;
        b[j] = (xv >= t0 && xv < t1) ? 1.0f : 0.0f;
    }
#pragma unroll
    for (int k = 1; k <= 3; ++k) {
        float inv = 1.0f / ((float)k * h);
#pragma unroll
        for (int j = 0; j + k < 11; ++j) {
            float tj    = (float)(j - 3) * h;          // grid[j]
            float tjk1  = (float)(j + k - 2) * h;      // grid[j+k+1]
            b[j] = (xv - tj) * inv * b[j] + (tjk1 - xv) * inv * b[j + 1];
        }
    }
#pragma unroll
    for (int j = 0; j < COEFF; ++j) bb[j] = b[j];
}

// ---------------- staged-panel WMMA: KP/32 k-steps, 2x4 tiles/wave ----------------
__device__ __forceinline__ void panel_mma(const half_t* As, const half_t* Bs,
                                          int aRow, int bCol, int l16, int lHi,
                                          v8f acc[2][4]) {
#pragma unroll
    for (int ks = 0; ks < KP / 32; ++ks) {
        v16h afr[2];
#pragma unroll
        for (int mt = 0; mt < 2; ++mt) {
            // A 16x32 f16 layout: lane m: K0..7 & K16..23 ; lane m+16: K8..15 & K24..31
            const half_t* p = &As[(aRow + mt * 16 + l16) * KPAD + ks * 32 + lHi * 8];
            v8h lo = *(const v8h*)p;
            v8h hi = *(const v8h*)(p + 16);
            afr[mt] = __builtin_shufflevector(lo, hi, 0, 1, 2, 3, 4, 5, 6, 7,
                                              8, 9, 10, 11, 12, 13, 14, 15);
        }
#pragma unroll
        for (int nt = 0; nt < 4; ++nt) {
            // B 32x16 f16 layout: lane n: K0..7 & K16..23 of col n ; lane n+16: K8..15 & K24..31
            const half_t* p = &Bs[(bCol + nt * 16 + l16) * KPAD + ks * 32 + lHi * 8];
            v8h lo = *(const v8h*)p;
            v8h hi = *(const v8h*)(p + 16);
            v16h bfr = __builtin_shufflevector(lo, hi, 0, 1, 2, 3, 4, 5, 6, 7,
                                               8, 9, 10, 11, 12, 13, 14, 15);
#pragma unroll
            for (int mt = 0; mt < 2; ++mt) {
                acc[mt][nt] = __builtin_amdgcn_wmma_f32_16x16x32_f16(
                    false, afr[mt], false, bfr, (short)0, acc[mt][nt], false, false);
            }
        }
    }
}

// ---------------- phase A staging: silu(x) tile + base_weight tile ----------------
__device__ __forceinline__ void loadA_global(const float* __restrict__ x,
                                             const float* __restrict__ bw,
                                             int rowBase, int colBase, int i0,
                                             int tid, v4f ra[8], v4f rb[8]) {
#pragma unroll
    for (int it = 0; it < 8; ++it) {
        int idx = (it * NTHREADS + tid) * 4;
        int r = idx / KP, k = idx % KP;
        ra[it] = *(const v4f*)(x  + (rowBase + r) * IN_DIM + i0 + k);
        rb[it] = *(const v4f*)(bw + (colBase + r) * IN_DIM + i0 + k);
    }
}

__device__ __forceinline__ void storeA_lds(half_t* As, half_t* Bs, int tid,
                                           const v4f ra[8], const v4f rb[8]) {
#pragma unroll
    for (int it = 0; it < 8; ++it) {
        int idx = (it * NTHREADS + tid) * 4;
        int r = idx / KP, k = idx % KP;
        v4h ha, hb;
#pragma unroll
        for (int e = 0; e < 4; ++e) {
            ha[e] = (half_t)silu_f(ra[it][e]);
            hb[e] = (half_t)rb[it][e];
        }
        *(v4h*)(&As[r * KPAD + k]) = ha;
        *(v4h*)(&Bs[r * KPAD + k]) = hb;
    }
}

// ------------- phase B staging: x values for bases + spline_weight tile -----------
__device__ __forceinline__ void loadB_global(const float* __restrict__ x,
                                             const float* __restrict__ sw,
                                             int rowBase, int colBase, int i0,
                                             int tid, float xr[4], v4f rb[8]) {
#pragma unroll
    for (int it = 0; it < 4; ++it) {
        int idx = it * NTHREADS + tid;
        int r = idx >> 3, ii = idx & 7;
        xr[it] = x[(rowBase + r) * IN_DIM + i0 + ii];
    }
#pragma unroll
    for (int it = 0; it < 8; ++it) {
        int idx = (it * NTHREADS + tid) * 4;
        int c = idx / KP, k = idx % KP;
        rb[it] = *(const v4f*)(sw + (colBase + c) * (IN_DIM * COEFF) + i0 * COEFF + k);
    }
}

__device__ __forceinline__ void storeB_lds(half_t* As, half_t* Bs, int tid,
                                           const float xr[4], const v4f rb[8]) {
#pragma unroll
    for (int it = 0; it < 4; ++it) {
        int idx = it * NTHREADS + tid;
        int r = idx >> 3, ii = idx & 7;
        float bb[COEFF];
        eval_bases(xr[it], bb);
        v8h hv;
#pragma unroll
        for (int g = 0; g < COEFF; ++g) hv[g] = (half_t)bb[g];
        *(v8h*)(&As[r * KPAD + ii * COEFF]) = hv;
    }
#pragma unroll
    for (int it = 0; it < 8; ++it) {
        int idx = (it * NTHREADS + tid) * 4;
        int c = idx / KP, k = idx % KP;
        v4h hv;
#pragma unroll
        for (int e = 0; e < 4; ++e) hv[e] = (half_t)rb[it][e];
        *(v4h*)(&Bs[c * KPAD + k]) = hv;
    }
}

__global__ __launch_bounds__(NTHREADS, 1)
void kan_fused_wmma(const float* __restrict__ x,
                    const float* __restrict__ bw,   // (OUT, IN)
                    const float* __restrict__ sw,   // (OUT, IN, 8)
                    float* __restrict__ out) {      // (B, OUT)
    __shared__ __align__(16) half_t As[2][BM * KPAD];
    __shared__ __align__(16) half_t Bs[2][BN * KPAD];

    const int tid   = threadIdx.x;
    const int lane  = tid & 31;
    const int wave  = tid >> 5;
    const int waveM = wave & 3;    // 0..3
    const int waveN = wave >> 2;   // 0..1
    const int l16   = lane & 15;
    const int lHi   = lane >> 4;

    const int rowBase = blockIdx.y * BM;
    const int colBase = blockIdx.x * BN;

    const int aRow = waveM * 32;   // wave tile: 32 rows x 64 cols
    const int bCol = waveN * 64;

    v8f acc[2][4];
#pragma unroll
    for (int mt = 0; mt < 2; ++mt)
#pragma unroll
        for (int nt = 0; nt < 4; ++nt) acc[mt][nt] = v8f{};

    v4f   ra[8], rb[8];
    float xr[4];

    // =============== Phase A: silu(x) @ base_weight^T (pipelined) ===============
    loadA_global(x, bw, rowBase, colBase, 0, tid, ra, rb);
    for (int p = 0; p < NPA; ++p) {
        half_t* Ab = As[p & 1];
        half_t* Bb = Bs[p & 1];
        storeA_lds(Ab, Bb, tid, ra, rb);
        __syncthreads();
        if (p + 1 < NPA) {
            loadA_global(x, bw, rowBase, colBase, (p + 1) * KP, tid, ra, rb);
        } else {
            // pre-load first phase-B panel so the pipeline never drains
            loadB_global(x, sw, rowBase, colBase, 0, tid, xr, rb);
        }
        panel_mma(Ab, Bb, aRow, bCol, l16, lHi, acc);
    }

    // ========= Phase B: bases(x) @ spline_weight^T, K = IN*8 (pipelined) =========
    for (int p = 0; p < NPB; ++p) {
        half_t* Ab = As[p & 1];
        half_t* Bb = Bs[p & 1];
        storeB_lds(Ab, Bb, tid, xr, rb);
        __syncthreads();
        if (p + 1 < NPB) {
            loadB_global(x, sw, rowBase, colBase, (p + 1) * (KP / COEFF), tid, xr, rb);
        }
        panel_mma(Ab, Bb, aRow, bCol, l16, lHi, acc);
    }

    // ================= Epilogue: f32 C/D layout stores =================
#pragma unroll
    for (int mt = 0; mt < 2; ++mt) {
#pragma unroll
        for (int nt = 0; nt < 4; ++nt) {
            int r0 = rowBase + aRow + mt * 16 + lHi * 8;  // lane<16: M=e ; lane>=16: M=8+e
            int c  = colBase + bCol + nt * 16 + l16;
#pragma unroll
            for (int e = 0; e < 8; ++e)
                out[(r0 + e) * OUT_DIM + c] = acc[mt][nt][e];
        }
    }
}

extern "C" void kernel_launch(void* const* d_in, const int* in_sizes, int n_in,
                              void* d_out, int out_size, void* d_ws, size_t ws_size,
                              hipStream_t stream) {
    const float* x  = (const float*)d_in[0];   // (8192, 1024)
    const float* bw = (const float*)d_in[1];   // (1024, 1024)
    const float* sw = (const float*)d_in[2];   // (1024, 1024, 8)
    float* out = (float*)d_out;                // (8192, 1024)

    dim3 grid(OUT_DIM / BN, B_DIM / BM);       // (8, 64)
    kan_fused_wmma<<<grid, NTHREADS, 0, stream>>>(x, bw, sw, out);
}
